// Model_57939108823205
// MI455X (gfx1250) — matrix operand order repaired and checked by reading, not compiled, not run on hardware
//
#include <hip/hip_runtime.h>
#include <math.h>

typedef __attribute__((ext_vector_type(16))) _Float16 v16h;
typedef __attribute__((ext_vector_type(8)))  _Float16 v8h;
typedef __attribute__((ext_vector_type(2)))  _Float16 v2h;
typedef __attribute__((ext_vector_type(8)))  float    v8f;

#define NB_B 8
#define SEQ 95
#define CIN 14
#define DM 256
#define NH 8
#define HD 32
#define DFF 512
#define EE 94
#define FF 47
#define H4 188      /* F*HSF */
#define MAXS 48
#define DECL 72
#define PREDL 24
#define ECHUNK 16
#define LAMBDA_C 0.01f
#define EPS_C 2.2204460492503131e-16f

__device__ inline float pe_val(int pos, int d) {
  int i2 = (d >> 1) * 2;
  float div = expf(-(float)i2 * (9.210340371976184f / 256.0f));
  float a = (float)pos * div;
  return (d & 1) ? cosf(a) : sinf(a);
}

// ---------------------------------------------------------------- WMMA GEMM
// C[e] = act(A[e] @ W[e] + bias[e]).
// Requirements (all call sites satisfy): M % 64 == 0, N % 128 == 0, K % 32 == 0.
// Block tile 64x128, 8 waves; each wave: 1 A fragment x 4 B tiles -> 4 accumulators.
__global__ __launch_bounds__(256) void gemm_ws_kernel(
    const float* __restrict__ A, long long sEA, int lda,
    const float* __restrict__ W, long long sEW,
    const float* __restrict__ bias, long long sEb,
    float* __restrict__ O, long long sEO,
    int M, int N, int K, int act)
{
  __shared__ __align__(16) _Float16 la[64 * 40];
  __shared__ __align__(16) _Float16 lb[128 * 40];
  const int e = blockIdx.z;
  const float* Ae = A + (long long)e * sEA;
  const float* We = W + (long long)e * sEW;
  float* Oe = O + (long long)e * sEO;
  const int m0 = blockIdx.x * 64;
  const int n0 = blockIdx.y * 128;
  const int tid = threadIdx.x;
  const int lane = tid & 31;
  const int wave = tid >> 5;
  const int tm = wave >> 1;            // 0..3  (16-row slice)
  const int tn0 = (wave & 1) * 4;      // 0 or 4 (four 16-col tiles)

  v8f accs[4] = {};

  // A tile loader: 64 rows x 32 cols, 8 f32 per thread
  const int arow = tid >> 2;           // 0..63
  const int acol = (tid & 3) * 8;      // 0,8,16,24
  // B tile loader: 32 rows(k) x 128 cols(n); thread handles k-pair x 8 cols
  const int bkp = (tid >> 4) * 2;      // k = 0,2,...,30
  const int bnc = (tid & 15) * 8;      // n offset 0..120

  for (int k0 = 0; k0 < K; k0 += 32) {
    {  // A tile -> LDS f16 row-major [m][k]
      const float* src = Ae + (long long)(m0 + arow) * lda + (k0 + acol);
      _Float16* dst = &la[arow * 40 + acol];
      #pragma unroll
      for (int j = 0; j < 8; ++j) dst[j] = (_Float16)src[j];
    }
    {  // W tile -> LDS f16 transposed [n][k]; pack K-pairs into b32 stores
      const float* s0 = We + (long long)(k0 + bkp) * N + (n0 + bnc);
      const float* s1 = s0 + N;
      #pragma unroll
      for (int j = 0; j < 8; ++j) {
        v2h pk;
        pk.x = (_Float16)s0[j];
        pk.y = (_Float16)s1[j];
        *(v2h*)&lb[(bnc + j) * 40 + bkp] = pk;
      }
    }
    if (k0 + 32 < K) {
      __builtin_prefetch(Ae + (long long)(m0 + arow) * lda + (k0 + 32 + acol), 0, 1);
      __builtin_prefetch(We + (long long)(k0 + 32 + bkp) * N + (n0 + bnc), 0, 1);
    }
    __syncthreads();

    // A fragment: lane m=lane%16; K base 0 (lanes 0-15) / 8 (lanes 16-31); +16 upper half
    const int am = lane & 15;
    const int akb = (lane < 16) ? 0 : 8;
    const _Float16* pa = &la[(tm * 16 + am) * 40 + akb];
    v8h alo = *(const v8h*)(pa);
    v8h ahi = *(const v8h*)(pa + 16);
    v16h af = __builtin_shufflevector(alo, ahi, 0,1,2,3,4,5,6,7,8,9,10,11,12,13,14,15);

    // B fragments: lane n=lane%16; K base 0 (lanes 0-15) / 8 (lanes 16-31); +16 upper half
    const int bn = lane & 15;
    const int bkb = (lane < 16) ? 0 : 8;
    #pragma unroll
    for (int t = 0; t < 4; ++t) {
      const _Float16* pb = &lb[((tn0 + t) * 16 + bn) * 40 + bkb];
      v8h blo = *(const v8h*)(pb);
      v8h bhi = *(const v8h*)(pb + 16);
      v16h bf = __builtin_shufflevector(blo, bhi, 0,1,2,3,4,5,6,7,8,9,10,11,12,13,14,15);
      accs[t] = __builtin_amdgcn_wmma_f32_16x16x32_f16(false, af, false, bf, (short)0, accs[t], false, false);
    }
    __syncthreads();
  }

  // epilogue: C layout N=lane%16, M = vgpr + (lane<16?0:8)
  const int cn = lane & 15;
  const int cmb = (lane < 16) ? 0 : 8;
  #pragma unroll
  for (int t = 0; t < 4; ++t) {
    const int col = n0 + (tn0 + t) * 16 + cn;
    float bv = 0.0f;
    if (bias) bv = bias[(long long)e * sEb + col];
    #pragma unroll
    for (int v = 0; v < 8; ++v) {
      const int row = m0 + tm * 16 + cmb + v;
      float x = accs[t][v] + bv;
      if (act == 1) x = 0.5f * x * (1.0f + erff(x * 0.70710678118654752f));
      Oe[(long long)row * N + col] = x;
    }
  }
}

// ---------------------------------------------------------------- attention
__global__ __launch_bounds__(128) void attn_kernel(
    const float* __restrict__ Q, long long sEQ,
    const float* __restrict__ Km, long long sEK,
    const float* __restrict__ V, long long sEV,
    float* __restrict__ O, long long sEO,
    int Sq, int Skv, int causal, int e0, int maskEnc)
{
  __shared__ float ks[72 * 33];
  __shared__ float vs[72 * 33];
  __shared__ float probs[4][72];
  __shared__ float qs[4][32];
  const int e = blockIdx.y;
  const int b = blockIdx.x >> 3;
  const int h = blockIdx.x & 7;
  const int ge = e0 + e;
  int valid = Skv;
  if (maskEnc) { int p = ge + 2; int nP = (SEQ + p - 1) / p; valid = nP < Skv ? nP : Skv; }
  const float* Ke = Km + (long long)e * sEK + (long long)b * Skv * DM + h * HD;
  const float* Ve = V  + (long long)e * sEV + (long long)b * Skv * DM + h * HD;
  const float* Qe = Q  + (long long)e * sEQ + (long long)b * Sq  * DM + h * HD;
  float* Oe = O + (long long)e * sEO + (long long)b * Sq * DM + h * HD;
  const int tid = threadIdx.x;
  for (int idx = tid; idx < Skv * HD; idx += 128) {
    int t = idx >> 5, d = idx & 31;
    ks[t * 33 + d] = Ke[(long long)t * DM + d];
    vs[t * 33 + d] = Ve[(long long)t * DM + d];
  }
  __syncthreads();
  const int w = tid >> 5, lane = tid & 31;
  const float scale = 0.17677669529663687f; // 1/sqrt(32)
  const int nt = (Skv + 31) >> 5;
  for (int s = w; s < Sq; s += 4) {
    qs[w][lane] = Qe[(long long)s * DM + lane];
    float sc[3] = {-1e30f, -1e30f, -1e30f};
    for (int j = 0; j < nt; ++j) {
      int t = lane + (j << 5);
      if (t < Skv) {
        float acc = 0.0f;
        #pragma unroll
        for (int d = 0; d < 32; ++d) acc += qs[w][d] * ks[t * 33 + d];
        acc *= scale;
        if (causal && t > s) acc = -1e30f;
        if (t >= valid) acc = -1e30f;
        sc[j] = acc;
      }
    }
    float m = fmaxf(sc[0], fmaxf(sc[1], sc[2]));
    for (int off = 16; off > 0; off >>= 1) m = fmaxf(m, __shfl_xor(m, off, 32));
    float ssum = 0.0f;
    for (int j = 0; j < nt; ++j) {
      int t = lane + (j << 5);
      if (t < Skv) { float pv = expf(sc[j] - m); probs[w][t] = pv; ssum += pv; }
    }
    for (int off = 16; off > 0; off >>= 1) ssum += __shfl_xor(ssum, off, 32);
    float inv = 1.0f / ssum;
    float o = 0.0f;
    for (int t = 0; t < Skv; ++t) o += probs[w][t] * vs[t * 33 + lane];
    Oe[(long long)s * DM + lane] = o * inv;
  }
}

// ---------------------------------------------------------------- layernorm (out = LN(X + R))
__global__ __launch_bounds__(256) void ln_kernel(
    const float* __restrict__ X, long long sEX,
    const float* __restrict__ R, long long sER,
    const float* __restrict__ g, const float* __restrict__ bt,
    float* __restrict__ O, long long sEO)
{
  __shared__ float red[256];
  const int e = blockIdx.y;
  const long long row = blockIdx.x;
  const int d = threadIdx.x;
  float x = X[(long long)e * sEX + row * DM + d];
  if (R) x += R[(long long)e * sER + row * DM + d];
  red[d] = x; __syncthreads();
  for (int s2 = 128; s2 > 0; s2 >>= 1) { if (d < s2) red[d] += red[d + s2]; __syncthreads(); }
  float mean = red[0] * (1.0f / 256.0f);
  __syncthreads();
  float c = x - mean;
  red[d] = c * c; __syncthreads();
  for (int s2 = 128; s2 > 0; s2 >>= 1) { if (d < s2) red[d] += red[d + s2]; __syncthreads(); }
  float var = red[0] * (1.0f / 256.0f);
  float y = g[(long long)e * DM + d] * c / sqrtf(var + 1e-5f) + bt[(long long)e * DM + d];
  O[(long long)e * sEO + row * DM + d] = y;
}

// ---------------------------------------------------------------- patch embedding
struct VePtrs { const float* p[ECHUNK]; };

__global__ __launch_bounds__(256) void patch_embed_kernel(
    const float* __restrict__ x_enc, VePtrs ve, int e0, float* __restrict__ X)
{
  __shared__ float lx[SEQ * CIN];
  const int e = blockIdx.x;
  const int b = blockIdx.y;
  const int ge = e0 + e;
  const int p = ge + 2;
  const int nP = (SEQ + p - 1) / p;
  const int tid = threadIdx.x;
  for (int i = tid; i < SEQ * CIN; i += 256) lx[i] = x_enc[b * SEQ * CIN + i];
  __syncthreads();
  const float* W = ve.p[e];
  const int d = tid;
  for (int s = 0; s < MAXS; ++s) {
    float acc = 0.0f;
    if (s < nP) {
      acc = pe_val(s, d);
      for (int c = 0; c < CIN; ++c)
        for (int i = 0; i < p; ++i) {
          int t = s * p + i; if (t > SEQ - 1) t = SEQ - 1;   // edge pad
          acc += lx[t * CIN + c] * W[(c * p + i) * DM + d];
        }
    }
    X[((long long)e * (NB_B * MAXS) + b * MAXS + s) * DM + d] = acc;
  }
}

// ---------------------------------------------------------------- decoder conv embedding
__global__ __launch_bounds__(256) void dec_embed_kernel(
    const float* __restrict__ x_dec, const float* __restrict__ Kc, float* __restrict__ dx)
{
  const int r = blockIdx.x;
  const int b = r / DECL, s = r % DECL;
  const int d = threadIdx.x;
  const int prev = (s == 0) ? DECL - 1 : s - 1;
  const int next = (s == DECL - 1) ? 0 : s + 1;
  float acc = pe_val(s, d);
  for (int c = 0; c < CIN; ++c) {
    acc += x_dec[(b * DECL + prev) * CIN + c] * Kc[(0 * CIN + c) * DM + d];
    acc += x_dec[(b * DECL + s)    * CIN + c] * Kc[(1 * CIN + c) * DM + d];
    acc += x_dec[(b * DECL + next) * CIN + c] * Kc[(2 * CIN + c) * DM + d];
  }
  dx[(long long)r * DM + d] = acc;
}

// ---------------------------------------------------------------- FFT gating / top-k router
__global__ __launch_bounds__(256) void gates_kernel(
    const float* __restrict__ x_enc,
    const float* __restrict__ w1, const float* __restrict__ b1,
    const float* __restrict__ w2, const float* __restrict__ b2,
    float* __restrict__ gates)
{
  __shared__ float Xr[CIN * FF], Xi[CIN * FF];
  __shared__ float o1r[CIN * H4], o1i[CIN * H4];
  __shared__ float wv[EE];
  const int b = blockIdx.x;
  const int tid = threadIdx.x;
  const float TWOPI = 6.283185307179586f;
  for (int idx = tid; idx < CIN * FF; idx += 256) {
    int c = idx / FF; int k = idx % FF + 1;
    float sr = 0.0f, si = 0.0f;
    for (int t = 0; t < SEQ; ++t) {
      int r = (k * t) % SEQ;
      float ang = -TWOPI * (float)r / (float)SEQ;
      float x = x_enc[(b * SEQ + t) * CIN + c];
      sr += x * cosf(ang); si += x * sinf(ang);
    }
    Xr[idx] = sr; Xi[idx] = si;
  }
  __syncthreads();
  for (int idx = tid; idx < CIN * H4; idx += 256) {
    int c = idx / H4; int j = idx % H4;
    float ar = b1[j], ai = b1[H4 + j];
    for (int f = 0; f < FF; ++f) {
      float xr = Xr[c * FF + f], xi = Xi[c * FF + f];
      float wa = w1[f * H4 + j];
      float wb = w1[FF * H4 + f * H4 + j];
      ar += xr * wa - xi * wb;
      ai += xi * wa + xr * wb;
    }
    o1r[idx] = ar > 0.0f ? ar : 0.0f;
    o1i[idx] = ai > 0.0f ? ai : 0.0f;
  }
  __syncthreads();
  if (tid < EE) {
    const int ei = tid;
    float acc = 0.0f;
    for (int c = 0; c < CIN; ++c) {
      float ar = b2[ei], ai = b2[EE + ei];
      for (int j = 0; j < H4; ++j) {
        float pr = o1r[c * H4 + j], pi = o1i[c * H4 + j];
        float wa = w2[j * EE + ei];
        float wb = w2[H4 * EE + j * EE + ei];
        ar += pr * wa - pi * wb;
        ai += pi * wa + pr * wb;
      }
      ar = (ar > LAMBDA_C) ? ar - LAMBDA_C : ((ar < -LAMBDA_C) ? ar + LAMBDA_C : 0.0f);
      ai = (ai > LAMBDA_C) ? ai - LAMBDA_C : ((ai < -LAMBDA_C) ? ai + LAMBDA_C : 0.0f);
      acc += sqrtf(ar * ar + ai * ai);
    }
    wv[ei] = acc * (1.0f / (float)CIN);
    gates[b * EE + ei] = 0.0f;
  }
  __syncthreads();
  if (tid == 0) {
    int sel[5]; float val[5];
    bool used[EE];
    for (int i = 0; i < EE; ++i) used[i] = false;
    for (int r = 0; r < 5; ++r) {
      float best = -1e30f; int bi = 0;
      for (int e2 = 0; e2 < EE; ++e2)
        if (!used[e2] && wv[e2] > best) { best = wv[e2]; bi = e2; }
      used[bi] = true; sel[r] = bi; val[r] = best;
    }
    float m = val[0];
    for (int r = 1; r < 5; ++r) m = fmaxf(m, val[r]);
    float sum = 0.0f;
    for (int r = 0; r < 5; ++r) { val[r] = expf(val[r] - m); sum += val[r]; }
    for (int r = 0; r < 5; ++r) gates[b * EE + sel[r]] = val[r] / sum;
  }
}

// ---------------------------------------------------------------- combine / final
__global__ void zero_kernel(float* p, long long n) {
  long long i = (long long)blockIdx.x * 256 + threadIdx.x;
  if (i < n) p[i] = 0.0f;
}

__global__ __launch_bounds__(256) void combine_kernel(
    const float* __restrict__ Xd, const float* __restrict__ gates,
    float* __restrict__ comb, int e0, int ecnt)
{
  const int r = blockIdx.x;           // b*24 + l
  const int b = r / PREDL, l = r % PREDL;
  const int d = threadIdx.x;
  float acc = 0.0f;
  for (int e = 0; e < ecnt; ++e) {
    float gv = gates[b * EE + e0 + e];
    float xv = Xd[((long long)e * (NB_B * DECL) + b * DECL + (MAXS + l)) * DM + d];
    acc += expf(xv) * gv;
  }
  comb[(long long)r * DM + d] += acc;
}

__global__ __launch_bounds__(256) void final_kernel(
    const float* __restrict__ comb, const float* __restrict__ pw,
    const float* __restrict__ pb, float* __restrict__ out)
{
  __shared__ float red[256];
  const int r = blockIdx.x; const int d = threadIdx.x;
  float v = comb[(long long)r * DM + d];
  if (v == 0.0f) v = EPS_C;
  red[d] = logf(v) * pw[d];
  __syncthreads();
  for (int s2 = 128; s2 > 0; s2 >>= 1) { if (d < s2) red[d] += red[d + s2]; __syncthreads(); }
  if (d == 0) {
    float val = red[0] + pb[0];
    for (int c = 0; c < CIN; ++c) out[r * CIN + c] = val;
  }
}

// ---------------------------------------------------------------- host orchestration
extern "C" void kernel_launch(void* const* d_in, const int* in_sizes, int n_in,
                              void* d_out, int out_size, void* d_ws, size_t ws_size,
                              hipStream_t stream)
{
  (void)in_sizes; (void)n_in; (void)out_size; (void)ws_size;
  auto F32 = [&](int i) { return (const float*)d_in[i]; };
  const float* x_enc = F32(0);
  const float* x_dec = F32(2);
  const float* w1 = F32(4); const float* b1 = F32(5);
  const float* w2 = F32(6); const float* b2 = F32(7);
  const float* dec_conv = F32(102);
  const float* encWq = F32(103); const float* encbq = F32(104);
  const float* encWk = F32(105); const float* encbk = F32(106);
  const float* encWv = F32(107); const float* encbv = F32(108);
  const float* encWo = F32(109); const float* encbo = F32(110);
  const float* encW1 = F32(111); const float* encbf1 = F32(112);
  const float* encW2 = F32(113); const float* encbf2 = F32(114);
  const float* encg1 = F32(115); const float* encb1v = F32(116);
  const float* encg2 = F32(117); const float* encb2v = F32(118);
  const float* encgf = F32(119); const float* encbfv = F32(120);
  const float* dsWq = F32(121); const float* dsbq = F32(122);
  const float* dsWk = F32(123); const float* dsbk = F32(124);
  const float* dsWv = F32(125); const float* dsbv = F32(126);
  const float* dsWo = F32(127); const float* dsbo = F32(128);
  const float* dcWq = F32(129); const float* dcbq = F32(130);
  const float* dcWk = F32(131); const float* dcbk = F32(132);
  const float* dcWv = F32(133); const float* dcbv = F32(134);
  const float* dcWo = F32(135); const float* dcbo = F32(136);
  const float* dW1 = F32(137); const float* dbf1 = F32(138);
  const float* dW2 = F32(139); const float* dbf2 = F32(140);
  const float* dg1 = F32(141); const float* db1v = F32(142);
  const float* dg2 = F32(143); const float* db2v = F32(144);
  const float* dg3 = F32(145); const float* db3v = F32(146);
  const float* dgf = F32(147); const float* dbfv = F32(148);
  const float* projW = F32(149); const float* projb = F32(150);

  size_t off = 0;
  auto carve = [&](size_t bytes) -> float* {
    float* p = (float*)((char*)d_ws + off);
    off += (bytes + 255) & ~(size_t)255;
    return p;
  };
  const long long encRows = (long long)NB_B * MAXS;   // 384
  const long long decRows = (long long)NB_B * DECL;   // 576
  float* gates = carve((size_t)NB_B * EE * 4);
  float* comb  = carve((size_t)NB_B * PREDL * DM * 4);
  float* dx    = carve((size_t)decRows * DM * 4);
  float* Xb    = carve((size_t)ECHUNK * encRows * DM * 4);
  float* encO  = carve((size_t)ECHUNK * encRows * DM * 4);
  float* Xd    = carve((size_t)ECHUNK * decRows * DM * 4);
  float* Qb    = carve((size_t)ECHUNK * decRows * DM * 4);
  float* Kb    = carve((size_t)ECHUNK * decRows * DM * 4);
  float* Vb    = carve((size_t)ECHUNK * decRows * DM * 4);
  float* Tb    = carve((size_t)ECHUNK * decRows * DM * 4);
  float* Ub    = carve((size_t)ECHUNK * decRows * DM * 4);
  float* Hb    = carve((size_t)ECHUNK * decRows * DFF * 4);

  const long long sENC = encRows * DM;
  const long long sDEC = decRows * DM;
  const long long sHN  = encRows * DFF;
  const long long sHD  = decRows * DFF;
  const long long sDD  = (long long)DM * DM;
  const long long sDF  = (long long)DM * DFF;

  { long long n = (long long)NB_B * PREDL * DM;
    zero_kernel<<<dim3((unsigned)((n + 255) / 256)), 256, 0, stream>>>(comb, n); }
  gates_kernel<<<NB_B, 256, 0, stream>>>(x_enc, w1, b1, w2, b2, gates);
  dec_embed_kernel<<<NB_B * DECL, 256, 0, stream>>>(x_dec, dec_conv, dx);

  for (int c0 = 0; c0 < EE; c0 += ECHUNK) {
    const int ec = (EE - c0 < ECHUNK) ? (EE - c0) : ECHUNK;
    VePtrs vp;
    for (int i = 0; i < ECHUNK; ++i) vp.p[i] = F32(8 + c0 + (i < ec ? i : 0));
    patch_embed_kernel<<<dim3(ec, NB_B), 256, 0, stream>>>(x_enc, vp, c0, Xb);

    // -------- encoder layer (per expert)
    dim3 ge1((unsigned)(encRows / 64), DM / 128, ec);
    gemm_ws_kernel<<<ge1, 256, 0, stream>>>(Xb, sENC, DM, encWq + (long long)c0 * sDD, sDD, encbq + (long long)c0 * DM, DM, Qb, sENC, (int)encRows, DM, DM, 0);
    gemm_ws_kernel<<<ge1, 256, 0, stream>>>(Xb, sENC, DM, encWk + (long long)c0 * sDD, sDD, encbk + (long long)c0 * DM, DM, Kb, sENC, (int)encRows, DM, DM, 0);
    gemm_ws_kernel<<<ge1, 256, 0, stream>>>(Xb, sENC, DM, encWv + (long long)c0 * sDD, sDD, encbv + (long long)c0 * DM, DM, Vb, sENC, (int)encRows, DM, DM, 0);
    attn_kernel<<<dim3(NB_B * NH, ec), 128, 0, stream>>>(Qb, sENC, Kb, sENC, Vb, sENC, Tb, sENC, MAXS, MAXS, 0, c0, 1);
    gemm_ws_kernel<<<ge1, 256, 0, stream>>>(Tb, sENC, DM, encWo + (long long)c0 * sDD, sDD, encbo + (long long)c0 * DM, DM, Ub, sENC, (int)encRows, DM, DM, 0);
    ln_kernel<<<dim3((unsigned)encRows, ec), 256, 0, stream>>>(Xb, sENC, Ub, sENC, encg1 + (long long)c0 * DM, encb1v + (long long)c0 * DM, Xb, sENC);
    dim3 ge2((unsigned)(encRows / 64), DFF / 128, ec);
    gemm_ws_kernel<<<ge2, 256, 0, stream>>>(Xb, sENC, DM, encW1 + (long long)c0 * sDF, sDF, encbf1 + (long long)c0 * DFF, DFF, Hb, sHN, (int)encRows, DFF, DM, 1);
    gemm_ws_kernel<<<ge1, 256, 0, stream>>>(Hb, sHN, DFF, encW2 + (long long)c0 * sDF, sDF, encbf2 + (long long)c0 * DM, DM, Ub, sENC, (int)encRows, DM, DFF, 0);
    ln_kernel<<<dim3((unsigned)encRows, ec), 256, 0, stream>>>(Xb, sENC, Ub, sENC, encg2 + (long long)c0 * DM, encb2v + (long long)c0 * DM, Xb, sENC);
    ln_kernel<<<dim3((unsigned)encRows, ec), 256, 0, stream>>>(Xb, sENC, (const float*)nullptr, 0LL, encgf + (long long)c0 * DM, encbfv + (long long)c0 * DM, encO, sENC);

    // -------- decoder layer (per expert)
    dim3 gd1((unsigned)(decRows / 64), DM / 128, ec);
    gemm_ws_kernel<<<gd1, 256, 0, stream>>>(dx, 0LL, DM, dsWq + (long long)c0 * sDD, sDD, dsbq + (long long)c0 * DM, DM, Qb, sDEC, (int)decRows, DM, DM, 0);
    gemm_ws_kernel<<<gd1, 256, 0, stream>>>(dx, 0LL, DM, dsWk + (long long)c0 * sDD, sDD, dsbk + (long long)c0 * DM, DM, Kb, sDEC, (int)decRows, DM, DM, 0);
    gemm_ws_kernel<<<gd1, 256, 0, stream>>>(dx, 0LL, DM, dsWv + (long long)c0 * sDD, sDD, dsbv + (long long)c0 * DM, DM, Vb, sDEC, (int)decRows, DM, DM, 0);
    attn_kernel<<<dim3(NB_B * NH, ec), 128, 0, stream>>>(Qb, sDEC, Kb, sDEC, Vb, sDEC, Tb, sDEC, DECL, DECL, 1, c0, 0);
    gemm_ws_kernel<<<gd1, 256, 0, stream>>>(Tb, sDEC, DM, dsWo + (long long)c0 * sDD, sDD, dsbo + (long long)c0 * DM, DM, Ub, sDEC, (int)decRows, DM, DM, 0);
    ln_kernel<<<dim3((unsigned)decRows, ec), 256, 0, stream>>>(Ub, sDEC, dx, 0LL, dg1 + (long long)c0 * DM, db1v + (long long)c0 * DM, Xd, sDEC);
    // cross attention
    gemm_ws_kernel<<<gd1, 256, 0, stream>>>(Xd, sDEC, DM, dcWq + (long long)c0 * sDD, sDD, dcbq + (long long)c0 * DM, DM, Qb, sDEC, (int)decRows, DM, DM, 0);
    gemm_ws_kernel<<<ge1, 256, 0, stream>>>(encO, sENC, DM, dcWk + (long long)c0 * sDD, sDD, dcbk + (long long)c0 * DM, DM, Kb, sENC, (int)encRows, DM, DM, 0);
    gemm_ws_kernel<<<ge1, 256, 0, stream>>>(encO, sENC, DM, dcWv + (long long)c0 * sDD, sDD, dcbv + (long long)c0 * DM, DM, Vb, sENC, (int)encRows, DM, DM, 0);
    attn_kernel<<<dim3(NB_B * NH, ec), 128, 0, stream>>>(Qb, sDEC, Kb, sENC, Vb, sENC, Tb, sDEC, DECL, MAXS, 0, c0, 1);
    gemm_ws_kernel<<<gd1, 256, 0, stream>>>(Tb, sDEC, DM, dcWo + (long long)c0 * sDD, sDD, dcbo + (long long)c0 * DM, DM, Ub, sDEC, (int)decRows, DM, DM, 0);
    ln_kernel<<<dim3((unsigned)decRows, ec), 256, 0, stream>>>(Xd, sDEC, Ub, sDEC, dg2 + (long long)c0 * DM, db2v + (long long)c0 * DM, Xd, sDEC);
    // FFN
    dim3 gd2((unsigned)(decRows / 64), DFF / 128, ec);
    gemm_ws_kernel<<<gd2, 256, 0, stream>>>(Xd, sDEC, DM, dW1 + (long long)c0 * sDF, sDF, dbf1 + (long long)c0 * DFF, DFF, Hb, sHD, (int)decRows, DFF, DM, 1);
    gemm_ws_kernel<<<gd1, 256, 0, stream>>>(Hb, sHD, DFF, dW2 + (long long)c0 * sDF, sDF, dbf2 + (long long)c0 * DM, DM, Ub, sDEC, (int)decRows, DM, DFF, 0);
    ln_kernel<<<dim3((unsigned)decRows, ec), 256, 0, stream>>>(Xd, sDEC, Ub, sDEC, dg3 + (long long)c0 * DM, db3v + (long long)c0 * DM, Xd, sDEC);
    ln_kernel<<<dim3((unsigned)decRows, ec), 256, 0, stream>>>(Xd, sDEC, (const float*)nullptr, 0LL, dgf + (long long)c0 * DM, dbfv + (long long)c0 * DM, Xd, sDEC);

    combine_kernel<<<NB_B * PREDL, 256, 0, stream>>>(Xd, gates, comb, c0, ec);
  }
  final_kernel<<<NB_B * PREDL, 256, 0, stream>>>(comb, projW, projb, (float*)d_out);
}
